// GaussionConvolution_F_78692390797701
// MI455X (gfx1250) — compile-verified, matrix operand order checked by reading, not run on hardware
//
#include <hip/hip_runtime.h>

#define N_NODES 100000
#define N_FEAT  512
#define UNITS   128
#define GAMMA   1.0f
#define KL_COEF 0.0005f

typedef __attribute__((ext_vector_type(16))) __bf16 v16bf;
typedef __attribute__((ext_vector_type(8)))  float  v8f;

#if defined(__has_builtin) && __has_builtin(__builtin_amdgcn_sched_barrier)
#define SCHED_FENCE() __builtin_amdgcn_sched_barrier(0)
#else
#define SCHED_FENCE() do {} while (0)
#endif

__device__ __forceinline__ void atom_add_f32(float* p, float v) {
  // Hardware global_atomic_add_f32 (no CAS loop).
  unsafeAtomicAdd(p, v);
}

__global__ void zero_kernel(float* __restrict__ out, int n) {
  const int n4 = n >> 2;
  float4* out4 = (float4*)out;
  for (int i = blockIdx.x * blockDim.x + threadIdx.x; i < n4; i += gridDim.x * blockDim.x)
    out4[i] = (float4){0.f, 0.f, 0.f, 0.f};
  // tail
  int t = (n4 << 2) + blockIdx.x * blockDim.x + threadIdx.x;
  if (t < n) out[t] = 0.0f;
}

// Fused GEMM (bf16 WMMA, f32 acc) + ELU/ReLU/attention epilogue + KL reduction.
// Block = 256 threads = 8 waves; each wave produces a 16x128 strip of h.
__global__ __launch_bounds__(256)
void gemm_act_kernel(const float* __restrict__ x, const float* __restrict__ kern,
                     float* __restrict__ ws0, float* __restrict__ ws1,
                     float* __restrict__ kl_out) {
  extern __shared__ char smem[];
  __bf16* ldsB = (__bf16*)smem;   // 128 fragments * 512 bf16 = 128 KB

  const int tid  = threadIdx.x;
  const int lane = tid & 31;
  const int wave = tid >> 5;
  const int half = lane >> 4;   // 0: lanes 0-15, 1: lanes 16-31
  const int l15  = lane & 15;

  // ---- Stage B (512x128 weights) into LDS as pre-swizzled bf16 WMMA fragments.
  // frag f = kstep*8 + nblk; lane L holds its 16 elements at ldsB[f*512 + L*16].
  // 16-bit B layout: lane gives N, VGPR v gives K pair: k = (v<4 ? 2v : 8+2v) + 8*half.
  for (int item = tid; item < 128 * 32; item += blockDim.x) {
    const int f = item >> 5, L = item & 31;
    const int kstep = f >> 3, nblk = f & 7;
    const int h8 = (L >> 4) * 8;
    const int n  = (L & 15) + nblk * 16;
    __bf16* dst = ldsB + (size_t)f * 512 + (size_t)L * 16;
    const float* src = kern + n;
#pragma unroll
    for (int v = 0; v < 8; ++v) {
      const int k = kstep * 32 + ((v < 4) ? 2 * v : 8 + 2 * v) + h8;
      dst[2 * v + 0] = (__bf16)src[(size_t)k * UNITS];
      dst[2 * v + 1] = (__bf16)src[(size_t)(k + 1) * UNITS];
    }
  }
  __syncthreads();

  const int rowbase = blockIdx.x * 128 + wave * 16;
  int rowm = rowbase + l15;
  if (rowm >= N_NODES) rowm = N_NODES - 1;          // clamp loads, keep EXEC all-ones
  const float* arow = x + (size_t)rowm * N_FEAT;

  v8f acc[8];
#pragma unroll
  for (int nb = 0; nb < 8; ++nb) acc[nb] = (v8f){0, 0, 0, 0, 0, 0, 0, 0};

  const int h8 = half * 8;
  const __bf16* ldsLane = ldsB + (size_t)lane * 16;
  for (int kstep = 0; kstep < 16; ++kstep) {
    const float* ap = arow + kstep * 32;
    __builtin_prefetch(ap + 32, 0, 1);              // global_prefetch_b8 next K-step
    // A fragment: global fp32 -> packed bf16
    v16bf afrag;
#pragma unroll
    for (int v = 0; v < 8; ++v) {
      const int kl = ((v < 4) ? 2 * v : 8 + 2 * v) + h8;
      float2 p2 = *(const float2*)(ap + kl);
      afrag[2 * v + 0] = (__bf16)p2.x;
      afrag[2 * v + 1] = (__bf16)p2.y;
    }
    // Preload ALL 8 B fragments of this K-step as a fenced batch: 16 ds_load_b128
    // pipeline under one s_wait_dscnt, then 8 WMMAs issue back-to-back.
    v16bf bfr[8];
    SCHED_FENCE();
#pragma unroll
    for (int nb = 0; nb < 8; ++nb)
      bfr[nb] = *(const v16bf*)(ldsLane + (size_t)(kstep * 8 + nb) * 512);
    SCHED_FENCE();
#pragma unroll
    for (int nb = 0; nb < 8; ++nb)
      acc[nb] = __builtin_amdgcn_wmma_f32_16x16x32_bf16(
          false, afrag, false, bfr[nb], (short)0, acc[nb], false, false);
    SCHED_FENCE();
  }

  // Epilogue: C/D layout -> lane gives N, VGPR r gives M = r + 8*half.
  float klacc = 0.f;
#pragma unroll
  for (int nb = 0; nb < 8; ++nb) {
    const int n = nb * 16 + l15;
#pragma unroll
    for (int r = 0; r < 8; ++r) {
      const int m = rowbase + r + h8;
      const float h    = acc[nb][r];
      const float mean = (h > 0.f) ? h : (__expf(h) - 1.f);   // elu
      const float var  = (h > 0.f) ? h : 0.f;                 // relu
      const float att  = __expf(-GAMMA * var);
      if (m < N_NODES) {
        klacc += 0.5f * (mean * mean + var - __logf(1e-8f + var) - 1.f);
        ws0[(size_t)m * UNITS + n] = mean * att;
        ws1[(size_t)m * UNITS + n] = var * att * att;
      }
    }
  }
#pragma unroll
  for (int off = 16; off > 0; off >>= 1) klacc += __shfl_down(klacc, off, 32);
  if (lane == 0) atom_add_f32(kl_out, klacc * (KL_COEF / (float)UNITS));
}

// One wave per edge (grid-stride): 4 features per lane, fp32 global atomics.
__global__ __launch_bounds__(256)
void spmm_kernel(const int* __restrict__ erow, const int* __restrict__ ecol,
                 const float* __restrict__ v0, const float* __restrict__ v1,
                 const float* __restrict__ ws0, const float* __restrict__ ws1,
                 float* __restrict__ mean_out, float* __restrict__ var_out, int n_edges) {
  const int lane   = threadIdx.x & 31;
  const int gw     = (blockIdx.x * blockDim.x + threadIdx.x) >> 5;
  const int nwaves = (gridDim.x * blockDim.x) >> 5;
  for (int e = gw; e < n_edges; e += nwaves) {
    const int   r  = erow[e];
    const int   c  = ecol[e];
    const float a0 = v0[e];
    const float a1 = v1[e];
    const float4 g0 = ((const float4*)(ws0 + (size_t)c * UNITS))[lane];
    const float4 g1 = ((const float4*)(ws1 + (size_t)c * UNITS))[lane];
    float* mo = mean_out + (size_t)r * UNITS + lane * 4;
    float* vo = var_out  + (size_t)r * UNITS + lane * 4;
    atom_add_f32(mo + 0, a0 * g0.x); atom_add_f32(mo + 1, a0 * g0.y);
    atom_add_f32(mo + 2, a0 * g0.z); atom_add_f32(mo + 3, a0 * g0.w);
    atom_add_f32(vo + 0, a1 * g1.x); atom_add_f32(vo + 1, a1 * g1.y);
    atom_add_f32(vo + 2, a1 * g1.z); atom_add_f32(vo + 3, a1 * g1.w);
  }
}

extern "C" void kernel_launch(void* const* d_in, const int* in_sizes, int n_in,
                              void* d_out, int out_size, void* d_ws, size_t ws_size,
                              hipStream_t stream) {
  const float* x    = (const float*)d_in[0];
  const float* kern = (const float*)d_in[1];
  const int*   erow = (const int*)d_in[2];
  const int*   ecol = (const int*)d_in[3];
  const float* a0   = (const float*)d_in[4];
  const float* a1   = (const float*)d_in[5];
  const int n_edges = in_sizes[2];

  float* out      = (float*)d_out;
  float* mean_out = out;
  float* var_out  = out + (size_t)N_NODES * UNITS;
  float* kl_out   = out + 2 * (size_t)N_NODES * UNITS;

  float* ws0 = (float*)d_ws;
  float* ws1 = ws0 + (size_t)N_NODES * UNITS;

  zero_kernel<<<2048, 256, 0, stream>>>(out, out_size);

  const int gblocks = (N_NODES + 127) / 128;
  gemm_act_kernel<<<gblocks, 256, 131072, stream>>>(x, kern, ws0, ws1, kl_out);

  spmm_kernel<<<4096, 256, 0, stream>>>(erow, ecol, a0, a1, ws0, ws1,
                                        mean_out, var_out, n_edges);
}
